// EncoderLayer_77610059039327
// MI455X (gfx1250) — compile-verified, matrix operand order checked by reading, not run on hardware
//
#include <hip/hip_runtime.h>
#include <hip/hip_bf16.h>
#include <stdint.h>

typedef __attribute__((ext_vector_type(16))) _Float16 v16h;
typedef __attribute__((ext_vector_type(8)))  float    v8f;

// Problem constants (match reference)
static constexpr int Bc   = 2;
static constexpr int Sc   = 2048;
static constexpr int Dc   = 512;
static constexpr int Hc   = 8;
static constexpr int DKc  = 64;
static constexpr int DFFc = 2048;

// ---------------------------------------------------------------------------
// Fragment loaders.  CDNA5 16-bit A-matrix 16x32 layout (wave32):
//   lanes 0-15  : row M=lane,    halves K = {0..7} then {16..23}
//   lanes 16-31 : row M=lane-16, halves K = {8..15} then {24..31}
// B fragments (from B^T stored row-major [N,K]) use the identical pattern.
// ---------------------------------------------------------------------------
__device__ __forceinline__ v16h load_frag_f16(const _Float16* __restrict__ base,
                                              int ld, int row0, int k0, int lane) {
    const int r = lane & 15, half = lane >> 4;
    const _Float16* p = base + (long long)(row0 + r) * ld + k0 + half * 8;
    union { uint4 u[2]; v16h v; } t;
    t.u[0] = *(const uint4*)(p);
    t.u[1] = *(const uint4*)(p + 16);
    return t.v;
}

__device__ __forceinline__ v16h load_frag_f32cvt(const float* __restrict__ base,
                                                 int ld, int row0, int k0, int lane) {
    const int r = lane & 15, half = lane >> 4;
    const float* p = base + (long long)(row0 + r) * ld + k0 + half * 8;
    float4 f0 = *(const float4*)(p);
    float4 f1 = *(const float4*)(p + 4);
    float4 f2 = *(const float4*)(p + 16);
    float4 f3 = *(const float4*)(p + 20);
    v16h v;
    v[0]=(_Float16)f0.x; v[1]=(_Float16)f0.y; v[2]=(_Float16)f0.z; v[3]=(_Float16)f0.w;
    v[4]=(_Float16)f1.x; v[5]=(_Float16)f1.y; v[6]=(_Float16)f1.z; v[7]=(_Float16)f1.w;
    v[8]=(_Float16)f2.x; v[9]=(_Float16)f2.y; v[10]=(_Float16)f2.z; v[11]=(_Float16)f2.w;
    v[12]=(_Float16)f3.x; v[13]=(_Float16)f3.y; v[14]=(_Float16)f3.z; v[15]=(_Float16)f3.w;
    return v;
}

// ---------------------------------------------------------------------------
// Batched NT WMMA GEMM: C[z] = A[z] (MxK, rm f16) * B[z]^T (B stored [N,K] rm f16)
// Workgroup = 8 waves -> 128(M) x 64(N) tile; wave = 4 stacked 16x16 tiles.
// Ping-pong double-buffered k-loop (2x32 unroll, no register rotation copies).
// REQUIRES: K % 64 == 0 (true for all call sites: 64, 512, 2048).
// ---------------------------------------------------------------------------
enum { M_PROJ_Q = 0, M_PROJ_K = 1, M_PROJ_V = 2, M_SCORES = 3,
       M_F16_BIAS_RELU = 4, M_F32_BIAS_RES = 5 };

__global__ __launch_bounds__(256)
void wmma_gemm_nt(const _Float16* __restrict__ A, const _Float16* __restrict__ Bm,
                  long long strideA, long long strideB,
                  int M, int N, int K, int lda, int ldb,
                  const float* __restrict__ bias, const float* __restrict__ res,
                  float* __restrict__ outF, _Float16* __restrict__ outH,
                  const unsigned char* __restrict__ mask, int mode)
{
    const int lane = threadIdx.x & 31;
    const int wave = threadIdx.x >> 5;
    const int half = lane >> 4;
    const int r    = lane & 15;
    const int z    = blockIdx.z;

    const _Float16* Az = A  + (long long)z * strideA;
    const _Float16* Bz = Bm + (long long)z * strideB;

    const int row0 = blockIdx.x * 128 + (wave >> 2) * 64;
    const int col0 = blockIdx.y * 64  + (wave & 3) * 16;

    v8f acc[4];
    const v8f zero = {0.f,0.f,0.f,0.f,0.f,0.f,0.f,0.f};
#pragma unroll
    for (int mt = 0; mt < 4; ++mt) acc[mt] = zero;

    // ---- ping-pong double buffer: set0 = k, set1 = k+32 ----
    v16h bf0 = load_frag_f16(Bz, ldb, col0, 0, lane);
    v16h af0[4];
#pragma unroll
    for (int mt = 0; mt < 4; ++mt)
        af0[mt] = load_frag_f16(Az, lda, row0 + mt * 16, 0, lane);
    v16h bf1 = load_frag_f16(Bz, ldb, col0, 32, lane);
    v16h af1[4];
#pragma unroll
    for (int mt = 0; mt < 4; ++mt)
        af1[mt] = load_frag_f16(Az, lda, row0 + mt * 16, 32, lane);

    for (int k0 = 64; k0 < K; k0 += 64) {
#pragma unroll
        for (int mt = 0; mt < 4; ++mt)
            acc[mt] = __builtin_amdgcn_wmma_f32_16x16x32_f16(
                false, af0[mt], false, bf0, (short)0, acc[mt], false, false);
        bf0 = load_frag_f16(Bz, ldb, col0, k0, lane);
#pragma unroll
        for (int mt = 0; mt < 4; ++mt)
            af0[mt] = load_frag_f16(Az, lda, row0 + mt * 16, k0, lane);
#pragma unroll
        for (int mt = 0; mt < 4; ++mt)
            acc[mt] = __builtin_amdgcn_wmma_f32_16x16x32_f16(
                false, af1[mt], false, bf1, (short)0, acc[mt], false, false);
        bf1 = load_frag_f16(Bz, ldb, col0, k0 + 32, lane);
#pragma unroll
        for (int mt = 0; mt < 4; ++mt)
            af1[mt] = load_frag_f16(Az, lda, row0 + mt * 16, k0 + 32, lane);
    }
#pragma unroll
    for (int mt = 0; mt < 4; ++mt)
        acc[mt] = __builtin_amdgcn_wmma_f32_16x16x32_f16(
            false, af0[mt], false, bf0, (short)0, acc[mt], false, false);
#pragma unroll
    for (int mt = 0; mt < 4; ++mt)
        acc[mt] = __builtin_amdgcn_wmma_f32_16x16x32_f16(
            false, af1[mt], false, bf1, (short)0, acc[mt], false, false);

    // Epilogue.  C-tile layout: VGPR i -> M = i + 8*half ; N = lane&15.
#pragma unroll
    for (int mt = 0; mt < 4; ++mt) {
#pragma unroll
        for (int i = 0; i < 8; ++i) {
            const int m = row0 + mt * 16 + half * 8 + i;
            const int n = col0 + r;
            if (m >= M || n >= N) continue;
            float v = acc[mt][i];
            if (mode == M_PROJ_Q || mode == M_PROJ_K) {
                v += bias[n];
                const int b = m >> 11, s = m & (Sc - 1);
                const int h = n >> 6,  d = n & 63;
                outH[((((long long)b * Hc + h) << 11) + s) * DKc + d] = (_Float16)v;
            } else if (mode == M_PROJ_V) {
                v += bias[n];
                const int b = m >> 11, s = m & (Sc - 1);
                const int h = n >> 6,  d = n & 63;
                outH[((((long long)b * Hc + h) * DKc + d) << 11) + s] = (_Float16)v;
            } else if (mode == M_SCORES) {
                v *= 0.125f;                       // 1/sqrt(64)
                const int b = z >> 3;              // z = b*H + h
                if (mask[((long long)b * Sc + m) * Sc + n]) v = -1e9f;
                outF[((long long)z * Sc + m) * Sc + n] = v;
            } else if (mode == M_F16_BIAS_RELU) {
                v += bias[n];
                v = fmaxf(v, 0.f);
                outH[(long long)m * N + n] = (_Float16)v;
            } else { // M_F32_BIAS_RES
                v += bias[n] + res[(long long)m * N + n];
                outF[(long long)m * N + n] = v;
            }
        }
    }
}

// ---------------------------------------------------------------------------
// context = attn (f32, converted per-fragment to f16) @ V   per (b,h).
// A = attn[z] [S,S] f32 ; B = Vt[z] [64,S] f16 ; out ctxh [B*S, H*64] f16.
// Same ping-pong structure; the f32->f16 cvt VALU co-executes with WMMAs.
// ---------------------------------------------------------------------------
__global__ __launch_bounds__(256)
void wmma_attn_v(const float* __restrict__ attn, const _Float16* __restrict__ Vt,
                 _Float16* __restrict__ ctxh)
{
    const int lane = threadIdx.x & 31;
    const int wave = threadIdx.x >> 5;
    const int half = lane >> 4;
    const int r    = lane & 15;
    const int z    = blockIdx.z;

    const float*    Ap = attn + (long long)z * Sc * Sc;
    const _Float16* Bp = Vt   + (long long)z * DKc * Sc;

    const int row0 = blockIdx.x * 128 + (wave >> 2) * 64;
    const int col0 = (wave & 3) * 16;            // N = 64 exactly

    v8f acc[4];
    const v8f zero = {0.f,0.f,0.f,0.f,0.f,0.f,0.f,0.f};
#pragma unroll
    for (int mt = 0; mt < 4; ++mt) acc[mt] = zero;

    v16h bf0 = load_frag_f16(Bp, Sc, col0, 0, lane);
    v16h af0[4];
#pragma unroll
    for (int mt = 0; mt < 4; ++mt)
        af0[mt] = load_frag_f32cvt(Ap, Sc, row0 + mt * 16, 0, lane);
    v16h bf1 = load_frag_f16(Bp, Sc, col0, 32, lane);
    v16h af1[4];
#pragma unroll
    for (int mt = 0; mt < 4; ++mt)
        af1[mt] = load_frag_f32cvt(Ap, Sc, row0 + mt * 16, 32, lane);

    for (int k0 = 64; k0 < Sc; k0 += 64) {
#pragma unroll
        for (int mt = 0; mt < 4; ++mt)
            acc[mt] = __builtin_amdgcn_wmma_f32_16x16x32_f16(
                false, af0[mt], false, bf0, (short)0, acc[mt], false, false);
        bf0 = load_frag_f16(Bp, Sc, col0, k0, lane);
#pragma unroll
        for (int mt = 0; mt < 4; ++mt)
            af0[mt] = load_frag_f32cvt(Ap, Sc, row0 + mt * 16, k0, lane);
#pragma unroll
        for (int mt = 0; mt < 4; ++mt)
            acc[mt] = __builtin_amdgcn_wmma_f32_16x16x32_f16(
                false, af1[mt], false, bf1, (short)0, acc[mt], false, false);
        bf1 = load_frag_f16(Bp, Sc, col0, k0 + 32, lane);
#pragma unroll
        for (int mt = 0; mt < 4; ++mt)
            af1[mt] = load_frag_f32cvt(Ap, Sc, row0 + mt * 16, k0 + 32, lane);
    }
#pragma unroll
    for (int mt = 0; mt < 4; ++mt)
        acc[mt] = __builtin_amdgcn_wmma_f32_16x16x32_f16(
            false, af0[mt], false, bf0, (short)0, acc[mt], false, false);
#pragma unroll
    for (int mt = 0; mt < 4; ++mt)
        acc[mt] = __builtin_amdgcn_wmma_f32_16x16x32_f16(
            false, af1[mt], false, bf1, (short)0, acc[mt], false, false);

    const int b = z >> 3, h = z & 7;
#pragma unroll
    for (int mt = 0; mt < 4; ++mt) {
#pragma unroll
        for (int i = 0; i < 8; ++i) {
            const int q = row0 + mt * 16 + half * 8 + i;
            const int d = col0 + r;
            ctxh[((long long)b * Sc + q) * Dc + h * DKc + d] = (_Float16)acc[mt][i];
        }
    }
}

// ---------------------------------------------------------------------------
// In-place row softmax over 2048 columns; one read + one write per element.
// ---------------------------------------------------------------------------
__global__ __launch_bounds__(256)
void softmax_rows(float* __restrict__ attn)
{
    __shared__ float red[256];
    float* p = attn + (long long)blockIdx.x * Sc;
    const int t = threadIdx.x;
    float v[8];
#pragma unroll
    for (int j = 0; j < 8; ++j) v[j] = p[t + j * 256];
    float mx = v[0];
#pragma unroll
    for (int j = 1; j < 8; ++j) mx = fmaxf(mx, v[j]);
    red[t] = mx; __syncthreads();
    for (int s = 128; s > 0; s >>= 1) { if (t < s) red[t] = fmaxf(red[t], red[t + s]); __syncthreads(); }
    mx = red[0]; __syncthreads();
    float sum = 0.f;
#pragma unroll
    for (int j = 0; j < 8; ++j) { v[j] = __expf(v[j] - mx); sum += v[j]; }
    red[t] = sum; __syncthreads();
    for (int s = 128; s > 0; s >>= 1) { if (t < s) red[t] += red[t + s]; __syncthreads(); }
    const float inv = 1.0f / red[0];
#pragma unroll
    for (int j = 0; j < 8; ++j) p[t + j * 256] = v[j] * inv;
}

// ---------------------------------------------------------------------------
// LayerNorm over rows of 512; writes f32 (and optional f16 copy for next GEMM).
// ---------------------------------------------------------------------------
__global__ __launch_bounds__(256)
void layernorm_rows(const float* __restrict__ y, const float* __restrict__ g,
                    const float* __restrict__ bb, float* __restrict__ outF,
                    _Float16* __restrict__ outH)
{
    __shared__ float red[256];
    const long long row = blockIdx.x;
    const float* p = y + row * Dc;
    const int t = threadIdx.x;
    const float a0 = p[t], a1 = p[t + 256];
    red[t] = a0 + a1; __syncthreads();
    for (int s = 128; s > 0; s >>= 1) { if (t < s) red[t] += red[t + s]; __syncthreads(); }
    const float mu = red[0] * (1.0f / Dc); __syncthreads();
    const float d0 = a0 - mu, d1 = a1 - mu;
    red[t] = d0 * d0 + d1 * d1; __syncthreads();
    for (int s = 128; s > 0; s >>= 1) { if (t < s) red[t] += red[t + s]; __syncthreads(); }
    const float inv = rsqrtf(red[0] * (1.0f / Dc) + 1e-5f);
    const float o0 = d0 * inv * g[t]       + bb[t];
    const float o1 = d1 * inv * g[t + 256] + bb[t + 256];
    outF[row * Dc + t]       = o0;
    outF[row * Dc + t + 256] = o1;
    if (outH) {
        outH[row * Dc + t]       = (_Float16)o0;
        outH[row * Dc + t + 256] = (_Float16)o1;
    }
}

// ---------------------------------------------------------------------------
// Conversions
// ---------------------------------------------------------------------------
__global__ __launch_bounds__(256)
void cvt_f32_to_f16(const float* __restrict__ in, _Float16* __restrict__ out, long long n)
{
    long long i = (long long)blockIdx.x * 256 + threadIdx.x;
    if (i < n) out[i] = (_Float16)in[i];
}

// W [K,N] f32  ->  WT [N,K] f16
__global__ __launch_bounds__(256)
void transpose_cvt_f16(const float* __restrict__ W, _Float16* __restrict__ WT,
                       int K, int N)
{
    long long idx = (long long)blockIdx.x * 256 + threadIdx.x;
    if (idx >= (long long)N * K) return;
    const int n = (int)(idx / K);
    const int k = (int)(idx - (long long)n * K);
    WT[idx] = (_Float16)W[(long long)k * N + n];
}

// ---------------------------------------------------------------------------
extern "C" void kernel_launch(void* const* d_in, const int* in_sizes, int n_in,
                              void* d_out, int out_size, void* d_ws, size_t ws_size,
                              hipStream_t stream)
{
    const float* X   = (const float*)d_in[0];
    const unsigned char* mask = (const unsigned char*)d_in[1]; // jax bool -> 1 byte
    const float* Wq  = (const float*)d_in[2];  const float* bq  = (const float*)d_in[3];
    const float* Wk  = (const float*)d_in[4];  const float* bk  = (const float*)d_in[5];
    const float* Wv  = (const float*)d_in[6];  const float* bv  = (const float*)d_in[7];
    const float* Wo  = (const float*)d_in[8];  const float* bo  = (const float*)d_in[9];
    const float* g1  = (const float*)d_in[10]; const float* be1 = (const float*)d_in[11];
    const float* W1  = (const float*)d_in[12]; const float* b1  = (const float*)d_in[13];
    const float* W2  = (const float*)d_in[14]; const float* b2  = (const float*)d_in[15];
    const float* g2  = (const float*)d_in[16]; const float* be2 = (const float*)d_in[17];

    float* outO = (float*)d_out;                          // [B,S,D]
    float* outA = outO + (size_t)Bc * Sc * Dc;            // [B,H,S,S]

    const int MT = Bc * Sc;                               // 4096 tokens

    char* w = (char*)d_ws;
    auto alloc = [&](size_t bytes) {
        void* p = (void*)w;
        w += (bytes + 255) & ~(size_t)255;
        return p;
    };
    _Float16* Xh    = (_Float16*)alloc((size_t)MT * Dc * 2);          // x in f16
    _Float16* WqT   = (_Float16*)alloc((size_t)Dc * Dc * 2);
    _Float16* WkT   = (_Float16*)alloc((size_t)Dc * Dc * 2);
    _Float16* WvT   = (_Float16*)alloc((size_t)Dc * Dc * 2);
    _Float16* WoT   = (_Float16*)alloc((size_t)Dc * Dc * 2);
    _Float16* W1T   = (_Float16*)alloc((size_t)DFFc * Dc * 2);
    _Float16* W2T   = (_Float16*)alloc((size_t)Dc * DFFc * 2);
    _Float16* Qh    = (_Float16*)alloc((size_t)Bc * Hc * Sc * DKc * 2); // [B,H,S,64]
    _Float16* Kh    = (_Float16*)alloc((size_t)Bc * Hc * Sc * DKc * 2); // [B,H,S,64]
    _Float16* Vt    = (_Float16*)alloc((size_t)Bc * Hc * DKc * Sc * 2); // [B,H,64,S]
    _Float16* ctxh  = (_Float16*)alloc((size_t)MT * Dc * 2);
    float*    y1    = (float*)alloc((size_t)MT * Dc * 4);             // pre-LN1
    float*    aoutF = (float*)alloc((size_t)MT * Dc * 4);             // attn_out f32
    _Float16* aoutH = (_Float16*)alloc((size_t)MT * Dc * 2);          // attn_out f16
    _Float16* hbuf  = (_Float16*)alloc((size_t)MT * DFFc * 2);        // FFN hidden
    float*    y2    = (float*)alloc((size_t)MT * Dc * 4);             // pre-LN2
    (void)ws_size; (void)in_sizes; (void)n_in; (void)out_size;

    // --- stage 0: conversions -------------------------------------------------
    {
        long long n = (long long)MT * Dc;
        cvt_f32_to_f16<<<(unsigned)((n + 255) / 256), 256, 0, stream>>>(X, Xh, n);
    }
    transpose_cvt_f16<<<(Dc * Dc + 255) / 256, 256, 0, stream>>>(Wq, WqT, Dc, Dc);
    transpose_cvt_f16<<<(Dc * Dc + 255) / 256, 256, 0, stream>>>(Wk, WkT, Dc, Dc);
    transpose_cvt_f16<<<(Dc * Dc + 255) / 256, 256, 0, stream>>>(Wv, WvT, Dc, Dc);
    transpose_cvt_f16<<<(Dc * Dc + 255) / 256, 256, 0, stream>>>(Wo, WoT, Dc, Dc);
    transpose_cvt_f16<<<(Dc * DFFc + 255) / 256, 256, 0, stream>>>(W1, W1T, Dc, DFFc);
    transpose_cvt_f16<<<(Dc * DFFc + 255) / 256, 256, 0, stream>>>(W2, W2T, DFFc, Dc);

    // --- stage 1: Q/K/V projections (fused bias + head-layout stores) --------
    {
        dim3 grid(MT / 128, Dc / 64, 1);
        wmma_gemm_nt<<<grid, 256, 0, stream>>>(Xh, WqT, 0, 0, MT, Dc, Dc, Dc, Dc,
                                               bq, nullptr, nullptr, Qh, nullptr, M_PROJ_Q);
        wmma_gemm_nt<<<grid, 256, 0, stream>>>(Xh, WkT, 0, 0, MT, Dc, Dc, Dc, Dc,
                                               bk, nullptr, nullptr, Kh, nullptr, M_PROJ_K);
        wmma_gemm_nt<<<grid, 256, 0, stream>>>(Xh, WvT, 0, 0, MT, Dc, Dc, Dc, Dc,
                                               bv, nullptr, nullptr, Vt, nullptr, M_PROJ_V);
    }

    // --- stage 2: scores = QK^T / sqrt(dk), masked; then row softmax ----------
    {
        dim3 grid(Sc / 128, Sc / 64, Bc * Hc);
        wmma_gemm_nt<<<grid, 256, 0, stream>>>(Qh, Kh,
                                               (long long)Sc * DKc, (long long)Sc * DKc,
                                               Sc, Sc, DKc, DKc, DKc,
                                               nullptr, nullptr, outA, nullptr, mask, M_SCORES);
        softmax_rows<<<Bc * Hc * Sc, 256, 0, stream>>>(outA);
    }

    // --- stage 3: context = attn @ V ------------------------------------------
    {
        dim3 grid(Sc / 128, 1, Bc * Hc);
        wmma_attn_v<<<grid, 256, 0, stream>>>(outA, Vt, ctxh);
    }

    // --- stage 4: attn_out = LN1(context @ Wo + bo + x) ------------------------
    {
        dim3 grid(MT / 128, Dc / 64, 1);
        wmma_gemm_nt<<<grid, 256, 0, stream>>>(ctxh, WoT, 0, 0, MT, Dc, Dc, Dc, Dc,
                                               bo, X, y1, nullptr, nullptr, M_F32_BIAS_RES);
        layernorm_rows<<<MT, 256, 0, stream>>>(y1, g1, be1, aoutF, aoutH);
    }

    // --- stage 5: FFN ----------------------------------------------------------
    {
        dim3 grid1(MT / 128, DFFc / 64, 1);
        wmma_gemm_nt<<<grid1, 256, 0, stream>>>(aoutH, W1T, 0, 0, MT, DFFc, Dc, Dc, Dc,
                                                b1, nullptr, nullptr, hbuf, nullptr, M_F16_BIAS_RELU);
        dim3 grid2(MT / 128, Dc / 64, 1);
        wmma_gemm_nt<<<grid2, 256, 0, stream>>>(hbuf, W2T, 0, 0, MT, Dc, DFFc, DFFc, DFFc,
                                                b2, aoutF, y2, nullptr, nullptr, M_F32_BIAS_RES);
        layernorm_rows<<<MT, 256, 0, stream>>>(y2, g2, be2, outO, nullptr);
    }
}
